// Attention_70153995813271
// MI455X (gfx1250) — matrix operand order repaired and checked by reading, not compiled, not run on hardware
//
#include <hip/hip_runtime.h>
#include <hip/hip_bf16.h>
#include <math.h>

// Problem constants (match reference)
constexpr int kB = 4;      // batch
constexpr int kC = 128;    // channels
constexpr int kN = 4096;   // spatial 16*16*16
constexpr int kG = 32;     // groups

typedef __bf16 bf16;
typedef __attribute__((ext_vector_type(16))) __bf16 v16bf;
typedef __attribute__((ext_vector_type(8)))  __bf16 v8bf;
typedef __attribute__((ext_vector_type(8)))  float  v8f;

static __device__ __forceinline__ v8f wmma_bf16f32(v16bf a, v16bf b, v8f c) {
  // D = A(16x32 bf16) * B(32x16 bf16) + C(16x16 f32)
  return __builtin_amdgcn_wmma_f32_16x16x32_bf16(false, a, false, b, (short)0, c,
                                                 false, false);
}

static __device__ __forceinline__ v16bf cat8(v8bf a, v8bf b) {
  return __builtin_shufflevector(a, b, 0, 1, 2, 3, 4, 5, 6, 7, 8, 9, 10, 11, 12,
                                 13, 14, 15);
}

// A-fragment (16x32 bf16): lane reads K = kc*32 + half*16 + 0..15
// from a K-innermost row. Two contiguous 16B chunks.
static __device__ __forceinline__ v16bf loadA(const bf16* rowptr, int kc,
                                              int half) {
  v8bf c0 = *(const v8bf*)(rowptr + kc * 32 + (half << 4));
  v8bf c1 = *(const v8bf*)(rowptr + kc * 32 + 8 + (half << 4));
  return cat8(c0, c1);
}

// B-fragment (32x16 bf16): lane reads K = 16*half + e -> 16 contiguous bf16.
static __device__ __forceinline__ v16bf loadB16(const bf16* p) {
  v8bf c0 = *(const v8bf*)(p);
  v8bf c1 = *(const v8bf*)(p + 8);
  return cat8(c0, c1);
}

// Low 32 bits of a generic pointer to __shared__ = wave-relative LDS offset.
static __device__ __forceinline__ unsigned lds_offset(const void* p) {
  return (unsigned)(unsigned long long)p;
}

// CDNA5 async DMA: 16 bytes per lane, global -> LDS, tracked by ASYNCcnt.
static __device__ __forceinline__ void async_load_b128(unsigned lds_off,
                                                       const void* gptr) {
  asm volatile("global_load_async_to_lds_b128 %0, %1, off" ::"v"(lds_off),
               "v"((unsigned long long)gptr)
               : "memory");
}
static __device__ __forceinline__ void wait_asynccnt0() {
  asm volatile("s_wait_asynccnt 0" ::: "memory");
}

static __device__ __forceinline__ float rowmax16(float v) {
  v = fmaxf(v, __shfl_xor(v, 1, 16));
  v = fmaxf(v, __shfl_xor(v, 2, 16));
  v = fmaxf(v, __shfl_xor(v, 4, 16));
  v = fmaxf(v, __shfl_xor(v, 8, 16));
  return v;
}
static __device__ __forceinline__ float rowsum16(float v) {
  v += __shfl_xor(v, 1, 16);
  v += __shfl_xor(v, 2, 16);
  v += __shfl_xor(v, 4, 16);
  v += __shfl_xor(v, 8, 16);
  return v;
}

// ---------------------------------------------------------------------------
// Kernel 1: convert weights f32 -> bf16 (all row-major [o][c]; K innermost).
// ---------------------------------------------------------------------------
__global__ __launch_bounds__(256) void cvt_weights_kernel(
    const float* __restrict__ Wq, const float* __restrict__ Wk,
    const float* __restrict__ Wv, const float* __restrict__ Wo,
    bf16* __restrict__ wqb, bf16* __restrict__ wkb, bf16* __restrict__ wvb,
    bf16* __restrict__ wob) {
  int i = blockIdx.x * blockDim.x + threadIdx.x;
  if (i >= kC * kC) return;
  wqb[i] = (bf16)Wq[i];
  wkb[i] = (bf16)Wk[i];
  wvb[i] = (bf16)Wv[i];
  wob[i] = (bf16)Wo[i];
}

// ---------------------------------------------------------------------------
// Kernel 2: fused QKV projection.
//   q_t[b][n][c], k_t[b][n][c] (transposed, K-innermost for attention GEMMs),
//   v[b][c][n]   (normal, K-innermost for the apply GEMM).
// Block: 256 thr (8 waves), one (b, 64-wide n tile); x tile staged [n][c].
// ---------------------------------------------------------------------------
constexpr int X_STRIDE = kC + 8;  // padded LDS row stride (bank spread)

__global__ __launch_bounds__(256) void qkv_proj_kernel(
    const float* __restrict__ x, const bf16* __restrict__ wq,
    const bf16* __restrict__ wk, const bf16* __restrict__ wv,
    const float* __restrict__ bq, const float* __restrict__ bk,
    const float* __restrict__ bv, bf16* __restrict__ qt,
    bf16* __restrict__ kt, bf16* __restrict__ vmat) {
  __shared__ bf16 lds_x[64 * X_STRIDE];  // [n_local][c]
  const int b    = blockIdx.x / (kN / 64);
  const int n0   = (blockIdx.x % (kN / 64)) * 64;
  const int tid  = threadIdx.x;
  const int lane = tid & 31;
  const int wid  = tid >> 5;
  const int half = lane >> 4;
  const int col  = lane & 15;

  // Stage x tile: coalesced global read (n fast), transposed LDS write.
  for (int j = 0; j < (kC * 64) / 256; ++j) {
    int idx = tid + j * 256;
    int c = idx >> 6, n = idx & 63;
    lds_x[n * X_STRIDE + c] = (bf16)x[((size_t)b * kC + c) * kN + n0 + n];
  }
  __syncthreads();

  const int o0 = wid * 16;

  // A fragments for the 3 weight matrices (vector chunk loads).
  v16bf aq[4], ak[4], av[4];
  for (int kc = 0; kc < 4; ++kc) {
    aq[kc] = loadA(wq + (o0 + col) * kC, kc, half);
    ak[kc] = loadA(wk + (o0 + col) * kC, kc, half);
    av[kc] = loadA(wv + (o0 + col) * kC, kc, half);
  }

  for (int nt = 0; nt < 4; ++nt) {
    v16bf bfr[4];
    for (int kc = 0; kc < 4; ++kc)
      bfr[kc] = loadB16(&lds_x[(nt * 16 + col) * X_STRIDE + kc * 32 + half * 16]);
    v8f accq = {0, 0, 0, 0, 0, 0, 0, 0};
    v8f acck = {0, 0, 0, 0, 0, 0, 0, 0};
    v8f accv = {0, 0, 0, 0, 0, 0, 0, 0};
    for (int kc = 0; kc < 4; ++kc) {
      accq = wmma_bf16f32(aq[kc], bfr[kc], accq);
      acck = wmma_bf16f32(ak[kc], bfr[kc], acck);
      accv = wmma_bf16f32(av[kc], bfr[kc], accv);
    }
    const int n = n0 + nt * 16 + col;  // this lane's spatial index
    v8bf pq, pk;
#pragma unroll
    for (int r = 0; r < 8; ++r) {
      int o = o0 + r + (half << 3);
      pq[r] = (bf16)(accq[r] + bq[o]);
      pk[r] = (bf16)(acck[r] + bk[o]);
      vmat[((size_t)b * kC + o) * kN + n] = (bf16)(accv[r] + bv[o]);
    }
    // 8 consecutive channels -> one 16B store each (transposed layouts).
    *(v8bf*)(qt + ((size_t)b * kN + n) * kC + o0 + (half << 3)) = pq;
    *(v8bf*)(kt + ((size_t)b * kN + n) * kC + o0 + (half << 3)) = pk;
  }
}

// ---------------------------------------------------------------------------
// Kernel 3: flash attention (no score scaling, matching reference).
//   S = Q K^T over C=128, online softmax over 4096 keys, O = P V^T.
//   h_t[b][n][c] bf16. K/V tiles staged via async global->LDS DMA.
// Block: 256 thr (8 waves) x 16 query rows each = 128 q rows / block.
// ---------------------------------------------------------------------------
constexpr int KT_STRIDE = kC + 8;  // lds_kt row stride (elements)
constexpr int V_STRIDE  = 32 + 8;  // lds_v row stride
constexpr int P_STRIDE  = 32 + 8;  // lds_p row stride

__global__ __launch_bounds__(256) void flash_attn_kernel(
    const bf16* __restrict__ qt, const bf16* __restrict__ kt,
    const bf16* __restrict__ vmat, bf16* __restrict__ ht) {
  __shared__ bf16 lds_kt[32 * KT_STRIDE];     // [m_local][c]
  __shared__ bf16 lds_v[kC * V_STRIDE];       // [c][m_local]
  __shared__ bf16 lds_p[8][16 * P_STRIDE];    // per-wave P [row][m_local]
  const int b    = blockIdx.x / (kN / 128);
  const int qblk = (blockIdx.x % (kN / 128)) * 128;
  const int tid  = threadIdx.x;
  const int lane = tid & 31;
  const int w    = tid >> 5;
  const int half = lane >> 4;
  const int col  = lane & 15;
  const int q0   = qblk + w * 16;

  // Q A-fragments (rows q0..q0+15, K = channel 0..127), vector loads.
  const bf16* qrow = qt + ((size_t)b * kN + q0 + col) * kC;
  v16bf aQ[4];
  for (int kc = 0; kc < 4; ++kc) aQ[kc] = loadA(qrow, kc, half);

  v8f accO[8];
#pragma unroll
  for (int i = 0; i < 8; ++i) accO[i] = (v8f){0, 0, 0, 0, 0, 0, 0, 0};
  float mrow[8], lrow[8];
#pragma unroll
  for (int r = 0; r < 8; ++r) { mrow[r] = -3.0e38f; lrow[r] = 0.0f; }

  for (int m0 = 0; m0 < kN; m0 += 32) {
    __syncthreads();
    // Async DMA staging: K tile 32x128 (16B chunks), V tile 128x32.
#pragma unroll
    for (int i = 0; i < 2; ++i) {
      int j = tid + i * 256;
      {  // k_t rows -> lds_kt[m][c]
        int m = j >> 4, ch = j & 15;
        async_load_b128(lds_offset(&lds_kt[m * KT_STRIDE + ch * 8]),
                        kt + ((size_t)b * kN + m0 + m) * kC + ch * 8);
      }
      {  // v rows -> lds_v[c][m]
        int c = j >> 2, ch = j & 3;
        async_load_b128(lds_offset(&lds_v[c * V_STRIDE + ch * 8]),
                        vmat + ((size_t)b * kC + c) * kN + m0 + ch * 8);
      }
    }
    wait_asynccnt0();
    __syncthreads();

    // Scores: 16 q-rows x 32 keys (B-fragments are contiguous 32B runs).
    v8f s0 = {0, 0, 0, 0, 0, 0, 0, 0};
    v8f s1 = {0, 0, 0, 0, 0, 0, 0, 0};
    for (int kc = 0; kc < 4; ++kc) {
      v16bf b0 = loadB16(&lds_kt[col * KT_STRIDE + kc * 32 + half * 16]);
      v16bf b1 = loadB16(&lds_kt[(16 + col) * KT_STRIDE + kc * 32 + half * 16]);
      s0 = wmma_bf16f32(aQ[kc], b0, s0);
      s1 = wmma_bf16f32(aQ[kc], b1, s1);
    }

    // Online softmax update per row
    float scale[8];
#pragma unroll
    for (int r = 0; r < 8; ++r) {
      float t = fmaxf(s0[r], s1[r]);
      t = rowmax16(t);
      float mnew = fmaxf(mrow[r], t);
      float sc = __expf(mrow[r] - mnew);
      float p0 = __expf(s0[r] - mnew);
      float p1 = __expf(s1[r] - mnew);
      float ps = rowsum16(p0 + p1);
      lrow[r] = lrow[r] * sc + ps;
      mrow[r] = mnew;
      scale[r] = sc;
      s0[r] = p0;
      s1[r] = p1;
    }
#pragma unroll
    for (int ct = 0; ct < 8; ++ct)
#pragma unroll
      for (int r = 0; r < 8; ++r) accO[ct][r] *= scale[r];

    // P: C/D layout -> A layout via per-wave LDS round trip.
#pragma unroll
    for (int r = 0; r < 8; ++r) {
      int row = r + (half << 3);
      lds_p[w][row * P_STRIDE + col]      = (bf16)s0[r];
      lds_p[w][row * P_STRIDE + 16 + col] = (bf16)s1[r];
    }
    v16bf aP = loadA(&lds_p[w][col * P_STRIDE], 0, half);

    // O += P (16x32) * V^T (32 x 128): B-fragments contiguous in lds_v rows.
#pragma unroll
    for (int ct = 0; ct < 8; ++ct) {
      v16bf bv_ = loadB16(&lds_v[(ct * 16 + col) * V_STRIDE + half * 16]);
      accO[ct] = wmma_bf16f32(aP, bv_, accO[ct]);
    }
  }

#pragma unroll
  for (int r = 0; r < 8; ++r) lrow[r] = 1.0f / lrow[r];
#pragma unroll
  for (int ct = 0; ct < 8; ++ct)
#pragma unroll
    for (int r = 0; r < 8; ++r) {
      int row = r + (half << 3);
      ht[((size_t)b * kN + q0 + row) * kC + ct * 16 + col] =
          (bf16)(accO[ct][r] * lrow[r]);
    }
}

// ---------------------------------------------------------------------------
// Kernel 4: output projection + bias + residual, y_t[b][n][c] f32, plus
// per-(b,group) sum / sumsq via shuffle-reduced atomics.
//   y_t[n][o] = sum_c h_t[n][c] * Wo[o][c] + bo[o] + x[b][o][n]
// Wo is row-major [o][c]: already K-innermost for B-fragments.
// ---------------------------------------------------------------------------
__global__ __launch_bounds__(256) void out_proj_kernel(
    const bf16* __restrict__ ht, const bf16* __restrict__ wob,
    const float* __restrict__ bo, const float* __restrict__ x,
    float* __restrict__ yt, float* __restrict__ stats) {
  const int b    = blockIdx.x / (kN / 128);
  const int nblk = (blockIdx.x % (kN / 128)) * 128;
  const int tid  = threadIdx.x;
  const int lane = tid & 31;
  const int w    = tid >> 5;
  const int half = lane >> 4;
  const int col  = lane & 15;
  const int n0   = nblk + w * 16;

  // A-fragments of h_t rows (K = channel), vector loads.
  const bf16* hrow = ht + ((size_t)b * kN + n0 + col) * kC;
  v16bf aH[4];
  for (int kc = 0; kc < 4; ++kc) aH[kc] = loadA(hrow, kc, half);

  for (int ot = 0; ot < 8; ++ot) {
    v8f acc = {0, 0, 0, 0, 0, 0, 0, 0};
    for (int kc = 0; kc < 4; ++kc) {
      v16bf bw = loadB16(wob + (ot * 16 + col) * kC + kc * 32 + half * 16);
      acc = wmma_bf16f32(aH[kc], bw, acc);
    }
    int o = ot * 16 + col;
    float bias = bo[o];
    float ps = 0.0f, ps2 = 0.0f;
#pragma unroll
    for (int r = 0; r < 8; ++r) {
      int n = n0 + r + (half << 3);
      float v = acc[r] + bias + x[((size_t)b * kC + o) * kN + n];
      yt[((size_t)b * kN + n) * kC + o] = v;
      ps += v;
      ps2 += v * v;
    }
    // Combine half-waves (same col), then the 4 cols of each channel group.
    ps += __shfl_xor(ps, 16, 32);  ps2 += __shfl_xor(ps2, 16, 32);
    ps += __shfl_xor(ps, 1, 32);   ps2 += __shfl_xor(ps2, 1, 32);
    ps += __shfl_xor(ps, 2, 32);   ps2 += __shfl_xor(ps2, 2, 32);
    if (half == 0 && (col & 3) == 0) {
      int g = o >> 2;
      atomicAdd(&stats[((size_t)b * kG + g) * 2 + 0], ps);
      atomicAdd(&stats[((size_t)b * kG + g) * 2 + 1], ps2);
    }
  }
}

// ---------------------------------------------------------------------------
// Kernel 5: GroupNorm + affine + Swish, write final [B][C][N] f32.
// ---------------------------------------------------------------------------
__global__ __launch_bounds__(256) void gn_swish_kernel(
    const float* __restrict__ yt, const float* __restrict__ stats,
    const float* __restrict__ gamma, const float* __restrict__ beta,
    float* __restrict__ out) {
  int i = blockIdx.x * 256 + threadIdx.x;
  if (i >= kB * kC * kN) return;
  int b = i / (kC * kN);
  int rem = i - b * (kC * kN);
  int c = rem / kN;
  int n = rem - c * kN;
  int g = c >> 2;  // kC/kG = 4 channels per group
  float s  = stats[((size_t)b * kG + g) * 2 + 0];
  float s2 = stats[((size_t)b * kG + g) * 2 + 1];
  const float inv = 1.0f / (float)(kN * (kC / kG));
  float mu = s * inv;
  float var = s2 * inv - mu * mu;
  float v = yt[((size_t)b * kN + n) * kC + c];
  float t = (v - mu) * rsqrtf(var + 1e-5f) * gamma[c] + beta[c];
  out[i] = t / (1.0f + __expf(-t));  // t * sigmoid(t)
}

// ---------------------------------------------------------------------------
extern "C" void kernel_launch(void* const* d_in, const int* in_sizes, int n_in,
                              void* d_out, int out_size, void* d_ws,
                              size_t ws_size, hipStream_t stream) {
  (void)in_sizes; (void)n_in; (void)out_size; (void)ws_size;
  const float* x     = (const float*)d_in[0];
  const float* Wq    = (const float*)d_in[1];
  const float* bq    = (const float*)d_in[2];
  const float* Wk    = (const float*)d_in[3];
  const float* bk    = (const float*)d_in[4];
  const float* Wv    = (const float*)d_in[5];
  const float* bv    = (const float*)d_in[6];
  const float* Wo    = (const float*)d_in[7];
  const float* bo    = (const float*)d_in[8];
  const float* gamma = (const float*)d_in[9];
  const float* beta  = (const float*)d_in[10];
  float* out = (float*)d_out;

  char* ws = (char*)d_ws;
  size_t off = 0;
  auto walloc = [&](size_t bytes) {
    char* p = ws + off;
    off = (off + bytes + 255) & ~(size_t)255;
    return p;
  };
  bf16* wqb  = (bf16*)walloc((size_t)kC * kC * 2);
  bf16* wkb  = (bf16*)walloc((size_t)kC * kC * 2);
  bf16* wvb  = (bf16*)walloc((size_t)kC * kC * 2);
  bf16* wob  = (bf16*)walloc((size_t)kC * kC * 2);
  bf16* qt   = (bf16*)walloc((size_t)kB * kN * kC * 2);
  bf16* kt   = (bf16*)walloc((size_t)kB * kN * kC * 2);
  bf16* vmat = (bf16*)walloc((size_t)kB * kN * kC * 2);
  bf16* htm  = (bf16*)walloc((size_t)kB * kN * kC * 2);
  float* yt  = (float*)walloc((size_t)kB * kN * kC * 4);
  float* st  = (float*)walloc((size_t)kB * kG * 2 * 4);

  hipMemsetAsync(st, 0, (size_t)kB * kG * 2 * 4, stream);

  cvt_weights_kernel<<<(kC * kC + 255) / 256, 256, 0, stream>>>(
      Wq, Wk, Wv, Wo, wqb, wkb, wvb, wob);
  qkv_proj_kernel<<<kB * (kN / 64), 256, 0, stream>>>(
      x, wqb, wkb, wvb, bq, bk, bv, qt, kt, vmat);
  flash_attn_kernel<<<kB * (kN / 128), 256, 0, stream>>>(qt, kt, vmat, htm);
  out_proj_kernel<<<kB * (kN / 128), 256, 0, stream>>>(htm, wob, bo, x, yt, st);
  gn_swish_kernel<<<(kB * kC * kN + 255) / 256, 256, 0, stream>>>(
      yt, st, gamma, beta, out);
}
